// LlamaAttention_5291399708750
// MI455X (gfx1250) — matrix operand order repaired and checked by reading, not compiled, not run on hardware
//
#include <hip/hip_runtime.h>
#include <hip/hip_bf16.h>

// ---------------------------------------------------------------------------
// LlamaAttention (B=1, S=2048, DMODEL=4096, H=32, HKV=8, DH=128) for gfx1250.
// All GEMMs + attention matmuls use v_wmma_f32_16x16x32_f16 (wave32 WMMA).
// GEMM: 128x128 block tile, 16 waves, 32x32 per wave => 4 WMMA / K-step / wave.
// ---------------------------------------------------------------------------

#define S_LEN   2048
#define DMODEL  4096
#define NH      32
#define NKV     8
#define DH      128
#define LTAB    20
#define RBKT    128
#define NEGBIG  (-1.0e9f)

typedef __attribute__((ext_vector_type(16))) _Float16 v16h;
typedef __attribute__((ext_vector_type(8)))  float    v8f;

union AF { v16h v; unsigned u[8]; };

// ===========================================================================
// 1) Sparse key mask: coll = sum_l q_probs[h,l,k_hard[h,l,t]];
//    score = allowed ? coll*v_norm : -inf ; top-M threshold via bitwise radix;
//    plus sink / sliding-window overrides.
// ===========================================================================
__global__ __launch_bounds__(256) void mask_kernel(
    const float* __restrict__ q_probs, const float* __restrict__ v_norm,
    const unsigned char* __restrict__ allowed, const int* __restrict__ k_hard,
    const int* __restrict__ sink_p, const int* __restrict__ window_p,
    const int* __restrict__ M_p, int* __restrict__ km)
{
    const int h = blockIdx.x;
    const int tid = threadIdx.x;
    const int T = S_LEN;
    __shared__ int cnt;

    unsigned mu[8];
    for (int j = 0; j < 8; ++j) {
        int t = tid + j * 256;
        const int*   kh = k_hard + ((size_t)h * LTAB) * T + t;
        const float* qp = q_probs + (size_t)h * LTAB * RBKT;
        float coll = 0.0f;
        #pragma unroll
        for (int l = 0; l < LTAB; ++l)
            coll += qp[l * RBKT + kh[(size_t)l * T]];
        float s = allowed[h * T + t] ? coll * v_norm[h * T + t] : -3.0e38f;
        unsigned b = __float_as_uint(s);
        mu[j] = (b & 0x80000000u) ? ~b : (b | 0x80000000u);
    }

    int Me = M_p[0]; if (Me > T) Me = T; if (Me < 0) Me = 0;

    unsigned thr = 0u;
    for (int bit = 31; bit >= 0; --bit) {
        unsigned cand = thr | (1u << bit);
        __syncthreads();
        if (tid == 0) cnt = 0;
        __syncthreads();
        int local = 0;
        for (int j = 0; j < 8; ++j) local += (mu[j] >= cand) ? 1 : 0;
        atomicAdd(&cnt, local);
        __syncthreads();
        if (cnt >= Me) thr = cand;
    }

    int sink = sink_p[0];     if (sink < 0) sink = 0;     if (sink > T) sink = T;
    int window = window_p[0]; if (window < 0) window = 0; if (window > T) window = T;
    int ws = T - window; if (ws < sink) ws = sink;

    for (int j = 0; j < 8; ++j) {
        int t = tid + j * 256;
        bool m = (Me > 0) && (mu[j] >= thr);
        if (t < sink) m = true;
        if (window > 0 && t >= ws) m = true;
        if (sink == 0 && window == 0 && t == T - 1) m = true;
        m = m && (allowed[h * T + t] != 0);
        km[h * T + t] = m ? 1 : 0;
    }
}

// ===========================================================================
// 2) WMMA GEMM: C[M][N] = A[M][K] (f32) @ B[K][N] (f32), f16 tensor path,
//    f32 accumulate. 128x128 block tile, 16 waves (512 thr), each wave 32x32
//    (2x2 fragments -> 4 v_wmma per 32-deep K step, A/B frag reuse).
// ===========================================================================
#define LDSP 40   // padded row stride (halfs) for LDS tiles

__global__ __launch_bounds__(512) void gemm_wmma(
    const float* __restrict__ A, const float* __restrict__ B,
    float* __restrict__ C, int M, int N, int K)
{
    __shared__ _Float16 As[128 * LDSP];   // [row][k]  128x32 (padded)
    __shared__ _Float16 Bs[128 * LDSP];   // [col][k]  128x32 (padded, transposed)

    const int tid  = threadIdx.x;
    const int lane = tid & 31;
    const int wid  = tid >> 5;
    const int wm   = wid >> 2;      // 0..3  (32-row macro tile)
    const int wn   = wid & 3;       // 0..3  (32-col macro tile)
    const int m0 = blockIdx.y * 128;
    const int n0 = blockIdx.x * 128;

    const int rl  = lane & 15;
    const int hg  = lane >> 4;
    const int kbA = hg * 8;
    const int kbB = hg * 16;

    v8f acc[2][2];
    #pragma unroll
    for (int i = 0; i < 2; ++i)
        #pragma unroll
        for (int j = 0; j < 2; ++j)
            #pragma unroll
            for (int r = 0; r < 8; ++r) acc[i][j][r] = 0.0f;

    // A staging: 128x32 floats = 1024 float4; 512 thr x 2 float4 (rows +0,+64)
    const int ar  = tid >> 3;
    const int ac  = (tid & 7) * 4;
    // B staging: 32x128 floats = 1024 float4; 512 thr x 2 float4 (k +0,+16)
    const int bk  = tid >> 5;
    const int bn  = (tid & 31) * 4;

    for (int k0 = 0; k0 < K; k0 += 32) {
        // ---- stage A tile (f32 -> f16) ----
        {
            const float4 av0 = *(const float4*)(A + (size_t)(m0 + ar) * K + k0 + ac);
            const float4 av1 = *(const float4*)(A + (size_t)(m0 + ar + 64) * K + k0 + ac);
            _Float16* d0 = &As[ar * LDSP + ac];
            d0[0] = (_Float16)av0.x; d0[1] = (_Float16)av0.y;
            d0[2] = (_Float16)av0.z; d0[3] = (_Float16)av0.w;
            _Float16* d1 = &As[(ar + 64) * LDSP + ac];
            d1[0] = (_Float16)av1.x; d1[1] = (_Float16)av1.y;
            d1[2] = (_Float16)av1.z; d1[3] = (_Float16)av1.w;
        }
        // ---- stage B tile transposed: Bs[n][k] ----
        {
            const float4 bv0 = *(const float4*)(B + (size_t)(k0 + bk) * N + n0 + bn);
            const float4 bv1 = *(const float4*)(B + (size_t)(k0 + bk + 16) * N + n0 + bn);
            Bs[(bn + 0) * LDSP + bk] = (_Float16)bv0.x;
            Bs[(bn + 1) * LDSP + bk] = (_Float16)bv0.y;
            Bs[(bn + 2) * LDSP + bk] = (_Float16)bv0.z;
            Bs[(bn + 3) * LDSP + bk] = (_Float16)bv0.w;
            Bs[(bn + 0) * LDSP + bk + 16] = (_Float16)bv1.x;
            Bs[(bn + 1) * LDSP + bk + 16] = (_Float16)bv1.y;
            Bs[(bn + 2) * LDSP + bk + 16] = (_Float16)bv1.z;
            Bs[(bn + 3) * LDSP + bk + 16] = (_Float16)bv1.w;
        }
        __syncthreads();

        AF a[2], b[2];
        #pragma unroll
        for (int t = 0; t < 2; ++t) {
            const _Float16* arow = &As[(wm * 32 + t * 16 + rl) * LDSP];
            #pragma unroll
            for (int i = 0; i < 4; ++i)
                a[t].u[i] = *(const unsigned*)&arow[kbA + 2 * i];
            #pragma unroll
            for (int i = 4; i < 8; ++i)
                a[t].u[i] = *(const unsigned*)&arow[16 + kbA + 2 * (i - 4)];
            const _Float16* brow = &Bs[(wn * 32 + t * 16 + rl) * LDSP];
            #pragma unroll
            for (int i = 0; i < 4; ++i)
                b[t].u[i] = *(const unsigned*)&brow[kbA + 2 * i];
            #pragma unroll
            for (int i = 4; i < 8; ++i)
                b[t].u[i] = *(const unsigned*)&brow[16 + kbA + 2 * (i - 4)];
        }
        #pragma unroll
        for (int i = 0; i < 2; ++i)
            #pragma unroll
            for (int j = 0; j < 2; ++j)
                acc[i][j] = __builtin_amdgcn_wmma_f32_16x16x32_f16(
                    false, a[i].v, false, b[j].v, (short)0, acc[i][j], false, false);
        __syncthreads();
    }

    #pragma unroll
    for (int i = 0; i < 2; ++i)
        #pragma unroll
        for (int j = 0; j < 2; ++j)
            #pragma unroll
            for (int r = 0; r < 8; ++r)
                C[(size_t)(m0 + wm * 32 + i * 16 + r + 8 * hg) * N
                  + n0 + wn * 32 + j * 16 + rl] = acc[i][j][r];
}

// ===========================================================================
// 3) RoPE + f16 conversion. Q -> [H][S][DH], K -> [HKV][S][DH],
//    V -> transposed [HKV][DH][S] (for contiguous PV B-fragments).
// ===========================================================================
__global__ __launch_bounds__(256) void rope_kernel(
    const float* __restrict__ Qf, const float* __restrict__ Kf,
    const float* __restrict__ Vf, const float* __restrict__ cosp,
    const float* __restrict__ sinp, _Float16* __restrict__ Qh,
    _Float16* __restrict__ Kh, _Float16* __restrict__ Vt)
{
    int idx = blockIdx.x * 256 + threadIdx.x;   // over S*H*DH
    int s = idx >> 12;          // / 4096
    int r = idx & 4095;
    int h = r >> 7;
    int d = r & 127;

    float c  = cosp[s * DH + d];
    float sn = sinp[s * DH + d];

    float qv = Qf[idx];
    float qr = (d < 64) ? -Qf[(size_t)s * DMODEL + h * DH + d + 64]
                        :  Qf[(size_t)s * DMODEL + h * DH + d - 64];
    Qh[((size_t)h * S_LEN + s) * DH + d] = (_Float16)(qv * c + qr * sn);

    if (h < NKV) {
        size_t kidx = (size_t)s * (NKV * DH) + h * DH + d;
        float kv = Kf[kidx];
        float kr = (d < 64) ? -Kf[kidx + 64] : Kf[kidx - 64];
        Kh[((size_t)h * S_LEN + s) * DH + d] = (_Float16)(kv * c + kr * sn);
        Vt[((size_t)h * DH + d) * S_LEN + s] = (_Float16)Vf[kidx];
    }
}

// ===========================================================================
// 4) Flash attention: one wave per (head, 16-query tile). Online softmax,
//    16 WMMAs per 32-key step (8 QK^T + 8 PV), DH=128 accumulators resident.
// ===========================================================================
__global__ __launch_bounds__(32) void attn_kernel(
    const _Float16* __restrict__ Qh, const _Float16* __restrict__ Kh,
    const _Float16* __restrict__ Vt, const int* __restrict__ km,
    float* __restrict__ Ao)
{
    const int qt   = blockIdx.x;
    const int h    = blockIdx.y;
    const int lane = threadIdx.x;
    const int q0   = qt * 16;
    const int kvh  = h >> 2;

    const int col = lane & 15;
    const int hg  = lane >> 4;
    const int kbA = hg * 8;
    const int kbB = hg * 16;

    __shared__ _Float16 Pb[16 * 36];

    // Preload Q fragments (A-layout, 4 x 32-K chunks covering DH=128)
    const _Float16* Qbase = Qh + ((size_t)h * S_LEN + q0 + col) * DH;
    AF qa[4];
    #pragma unroll
    for (int d = 0; d < 4; ++d) {
        #pragma unroll
        for (int i = 0; i < 4; ++i)
            qa[d].u[i] = *(const unsigned*)&Qbase[d * 32 + kbA + 2 * i];
        #pragma unroll
        for (int i = 4; i < 8; ++i)
            qa[d].u[i] = *(const unsigned*)&Qbase[d * 32 + 16 + kbA + 2 * (i - 4)];
    }

    v8f o[8];
    float m_run[8], l_run[8];
    #pragma unroll
    for (int dt = 0; dt < 8; ++dt)
        #pragma unroll
        for (int r = 0; r < 8; ++r) o[dt][r] = 0.0f;
    #pragma unroll
    for (int r = 0; r < 8; ++r) { m_run[r] = -__builtin_inff(); l_run[r] = 0.0f; }

    const _Float16* Kbase = Kh + (size_t)kvh * S_LEN * DH;
    const _Float16* Vbase = Vt + (size_t)kvh * DH * S_LEN;
    const int* kmh = km + h * S_LEN;
    const float scale = 0.08838834764831845f;   // 1/sqrt(128)

    for (int key0 = 0; key0 < q0 + 16; key0 += 32) {
        v8f sfr[2];
        // ---- QK^T for two 16-key subtiles ----
        #pragma unroll
        for (int kt = 0; kt < 2; ++kt) {
            v8f s;
            #pragma unroll
            for (int r = 0; r < 8; ++r) s[r] = 0.0f;
            const int key = key0 + kt * 16 + col;
            const _Float16* Krow = Kbase + (size_t)key * DH;
            #pragma unroll
            for (int d = 0; d < 4; ++d) {
                AF b;
                #pragma unroll
                for (int i = 0; i < 4; ++i)
                    b.u[i] = *(const unsigned*)&Krow[d * 32 + kbA + 2 * i];
                #pragma unroll
                for (int i = 4; i < 8; ++i)
                    b.u[i] = *(const unsigned*)&Krow[d * 32 + 16 + kbA + 2 * (i - 4)];
                s = __builtin_amdgcn_wmma_f32_16x16x32_f16(
                        false, qa[d].v, false, b.v, (short)0, s, false, false);
            }
            const int ok = kmh[key];
            #pragma unroll
            for (int r = 0; r < 8; ++r) {
                int q = q0 + r + 8 * hg;
                float sv = s[r] * scale;
                s[r] = (ok && key <= q) ? sv : NEGBIG;
            }
            sfr[kt] = s;
        }
        // ---- online softmax (row reductions across 16-lane half-waves) ----
        float mnew[8], alpha[8], rs[8];
        #pragma unroll
        for (int r = 0; r < 8; ++r) {
            float v = fmaxf(sfr[0][r], sfr[1][r]);
            v = fmaxf(v, __shfl_xor(v, 1, 16));
            v = fmaxf(v, __shfl_xor(v, 2, 16));
            v = fmaxf(v, __shfl_xor(v, 4, 16));
            v = fmaxf(v, __shfl_xor(v, 8, 16));
            mnew[r]  = fmaxf(m_run[r], v);
            alpha[r] = __expf(m_run[r] - mnew[r]);
        }
        #pragma unroll
        for (int r = 0; r < 8; ++r) {
            float p0 = __expf(sfr[0][r] - mnew[r]);
            float p1 = __expf(sfr[1][r] - mnew[r]);
            int row = r + 8 * hg;
            Pb[row * 36 + col]      = (_Float16)p0;   // P tile, A-frag layout source
            Pb[row * 36 + 16 + col] = (_Float16)p1;
            float ps = p0 + p1;
            ps += __shfl_xor(ps, 1, 16);
            ps += __shfl_xor(ps, 2, 16);
            ps += __shfl_xor(ps, 4, 16);
            ps += __shfl_xor(ps, 8, 16);
            rs[r] = ps;
        }
        #pragma unroll
        for (int r = 0; r < 8; ++r) {
            l_run[r] = l_run[r] * alpha[r] + rs[r];
            m_run[r] = mnew[r];
        }
        // single-wave workgroup: LDS stores -> loads; DS pipe is in-order,
        // fence the compiler + hw with an explicit dscnt wait (CDNA5 split cnt)
        asm volatile("s_wait_dscnt 0" ::: "memory");

        AF pa;
        const _Float16* prow = &Pb[col * 36];
        #pragma unroll
        for (int i = 0; i < 4; ++i)
            pa.u[i] = *(const unsigned*)&prow[kbA + 2 * i];
        #pragma unroll
        for (int i = 4; i < 8; ++i)
            pa.u[i] = *(const unsigned*)&prow[16 + kbA + 2 * (i - 4)];

        // ---- PV accumulation over DH=128 (8 dh tiles) ----
        #pragma unroll
        for (int dt = 0; dt < 8; ++dt) {
            #pragma unroll
            for (int r = 0; r < 8; ++r) o[dt][r] *= alpha[r];
            AF bv;
            const _Float16* vrow = Vbase + (size_t)(dt * 16 + col) * S_LEN + key0;
            #pragma unroll
            for (int i = 0; i < 4; ++i)
                bv.u[i] = *(const unsigned*)&vrow[kbA + 2 * i];
            #pragma unroll
            for (int i = 4; i < 8; ++i)
                bv.u[i] = *(const unsigned*)&vrow[16 + kbA + 2 * (i - 4)];
            o[dt] = __builtin_amdgcn_wmma_f32_16x16x32_f16(
                        false, pa.v, false, bv.v, (short)0, o[dt], false, false);
        }
        asm volatile("s_wait_dscnt 0" ::: "memory");
    }

    // normalize + store to [S][H*DH] for the output projection GEMM
    #pragma unroll
    for (int dt = 0; dt < 8; ++dt) {
        #pragma unroll
        for (int r = 0; r < 8; ++r) {
            int q = q0 + r + 8 * hg;
            Ao[(size_t)q * DMODEL + h * DH + dt * 16 + col] = o[dt][r] / l_run[r];
        }
    }
}

// ===========================================================================
// Launch
// ===========================================================================
extern "C" void kernel_launch(void* const* d_in, const int* in_sizes, int n_in,
                              void* d_out, int out_size, void* d_ws, size_t ws_size,
                              hipStream_t stream) {
    (void)in_sizes; (void)n_in; (void)out_size; (void)ws_size;

    const float* hs      = (const float*)d_in[0];
    const float* cosp    = (const float*)d_in[1];
    const float* sinp    = (const float*)d_in[2];
    const float* q_probs = (const float*)d_in[3];
    const float* v_norm  = (const float*)d_in[4];
    const unsigned char* allowed = (const unsigned char*)d_in[5];
    const int*   k_hard  = (const int*)d_in[6];
    const float* Wq      = (const float*)d_in[7];
    const float* Wk      = (const float*)d_in[8];
    const float* Wv      = (const float*)d_in[9];
    const float* Wo      = (const float*)d_in[10];
    const int*   sink_p  = (const int*)d_in[11];
    const int*   win_p   = (const int*)d_in[12];
    const int*   M_p     = (const int*)d_in[13];

    char* ws = (char*)d_ws;
    size_t off = 0;
    int* km = (int*)(ws + off);            off += (size_t)NH * S_LEN * 4;          // 256 KB
    float* Qf = (float*)(ws + off);        off += (size_t)S_LEN * DMODEL * 4;      // 32 MB
    float* Kf = (float*)(ws + off);        off += (size_t)S_LEN * NKV * DH * 4;    // 8 MB
    float* Vf = (float*)(ws + off);        off += (size_t)S_LEN * NKV * DH * 4;    // 8 MB
    _Float16* Qh = (_Float16*)(ws + off);  off += (size_t)S_LEN * DMODEL * 2;      // 16 MB
    _Float16* Kh = (_Float16*)(ws + off);  off += (size_t)S_LEN * NKV * DH * 2;    // 4 MB
    _Float16* Vt = (_Float16*)(ws + off);  off += (size_t)S_LEN * NKV * DH * 2;    // 4 MB
    float* Ao = Qf;   // Qf32 is dead after rope_kernel; reuse for attention out

    // 1) sparse key mask
    mask_kernel<<<NH, 256, 0, stream>>>(q_probs, v_norm, allowed, k_hard,
                                        sink_p, win_p, M_p, km);
    // 2) QKV projections (N multiples of 128: 4096 / 1024)
    gemm_wmma<<<dim3(DMODEL / 128, S_LEN / 128), 512, 0, stream>>>(
        hs, Wq, Qf, S_LEN, DMODEL, DMODEL);
    gemm_wmma<<<dim3((NKV * DH) / 128, S_LEN / 128), 512, 0, stream>>>(
        hs, Wk, Kf, S_LEN, NKV * DH, DMODEL);
    gemm_wmma<<<dim3((NKV * DH) / 128, S_LEN / 128), 512, 0, stream>>>(
        hs, Wv, Vf, S_LEN, NKV * DH, DMODEL);
    // 3) RoPE + f16 packing
    rope_kernel<<<(S_LEN * NH * DH) / 256, 256, 0, stream>>>(
        Qf, Kf, Vf, cosp, sinp, Qh, Kh, Vt);
    // 4) flash attention
    attn_kernel<<<dim3(S_LEN / 16, NH), 32, 0, stream>>>(Qh, Kh, Vt, km, Ao);
    // 5) output projection
    gemm_wmma<<<dim3(DMODEL / 128, S_LEN / 128), 512, 0, stream>>>(
        Ao, Wo, (float*)d_out, S_LEN, DMODEL, DMODEL);
}
